// SpatialAttention_77163382440727
// MI455X (gfx1250) — matrix operand order repaired and checked by reading, not compiled, not run on hardware
//
#include <hip/hip_runtime.h>
#include <hip/hip_bf16.h>
#include <stdint.h>

// CDNA5 / gfx1250 wave32 WMMA types
typedef __attribute__((ext_vector_type(16))) __bf16 v16bf;
typedef __attribute__((ext_vector_type(8)))  float  v8f;

#define N_SP  4096   // H*W
#define C_CH  256
#define CQ    32
#define BATCH 8

union V16U { v16bf v; uint4 u[2]; };

// ---------------------------------------------------------------------------
// Kernel 0: f32 -> bf16 weight preconversion (one-time, 160KB total)
// ---------------------------------------------------------------------------
__global__ __launch_bounds__(256)
void cvt_kernel(const float* __restrict__ src, __bf16* __restrict__ dst, int n) {
  const int i = blockIdx.x * 256 + threadIdx.x;
  if (i < n) dst[i] = static_cast<__bf16>(src[i]);
}

// ---------------------------------------------------------------------------
// Kernel 1: fused q/k/v 1x1-conv projections in bf16 WMMA, bias fused.
// One block per (batch, 16-col n-tile); 20 waves = 2 q-tiles + 2 k-tiles +
// 16 v-tiles. The 256x16 x strip is staged ONCE into LDS (transposed, bf16)
// so every wave's B-tiles are two ds_load_b128 per K-step.
// q/k stored transposed [B][N][32] bf16; v stored [B][C][N] bf16.
// ---------------------------------------------------------------------------
__global__ __launch_bounds__(640)
void proj_kernel(const float* __restrict__ x,
                 const __bf16* __restrict__ wbf,   // [wq 32x256][wk 32x256][wv 256x256]
                 const float* __restrict__ bq,
                 const float* __restrict__ bk,
                 const float* __restrict__ bv,
                 __bf16* __restrict__ qb,
                 __bf16* __restrict__ kbuf,
                 __bf16* __restrict__ vbuf) {
  __shared__ __align__(16) __bf16 xs[16 * C_CH];  // xs[n][c], 8KB

  const int b  = blockIdx.x / 256;
  const int nt = blockIdx.x % 256;
  const int n0 = nt * 16;

  // Cooperative stage: x[b][c][n0+n] -> xs[n*256+c] (bf16)
  const float* xb = x + (size_t)b * C_CH * N_SP;
  for (int idx = threadIdx.x; idx < 16 * C_CH; idx += 640) {
    const int c = idx >> 4;
    const int n = idx & 15;
    xs[n * C_CH + c] = static_cast<__bf16>(xb[(size_t)c * N_SP + n0 + n]);
  }
  __syncthreads();

  const int lane  = threadIdx.x & 31;
  const int wave  = threadIdx.x >> 5;          // 0..19
  const int half  = lane >> 4;
  const int row   = lane & 15;
  const int col   = lane & 15;

  const __bf16* w; const float* bias; int d0; int kind;
  if (wave < 2)      { w = wbf;             bias = bq; d0 = wave * 16;        kind = 0; }
  else if (wave < 4) { w = wbf + 32 * 256;  bias = bk; d0 = (wave - 2) * 16;  kind = 1; }
  else               { w = wbf + 64 * 256;  bias = bv; d0 = (wave - 4) * 16;  kind = 2; }

  v8f acc = {0.f,0.f,0.f,0.f,0.f,0.f,0.f,0.f};
  #pragma unroll
  for (int kk = 0; kk < C_CH; kk += 32) {
    // A tile: w[d0+row][kk..kk+31] bf16 row-major -> two 16B loads
    V16U a;
    const __bf16* wr = w + (size_t)(d0 + row) * C_CH + kk;
    a.u[0] = *reinterpret_cast<const uint4*>(wr + half * 16);
    a.u[1] = *reinterpret_cast<const uint4*>(wr + half * 16 + 8);
    // B tile: xs[col][kk + i + 16*half] contiguous -> two LDS 16B loads
    V16U bm;
    const __bf16* xp = &xs[col * C_CH + kk + half * 16];
    bm.u[0] = *reinterpret_cast<const uint4*>(xp);
    bm.u[1] = *reinterpret_cast<const uint4*>(xp + 8);
    acc = __builtin_amdgcn_wmma_f32_16x16x32_bf16(false, a.v, false, bm.v,
                                                  (short)0, acc, false, false);
  }

  if (kind < 2) {
    __bf16* dst = (kind == 0) ? qb : kbuf;
    #pragma unroll
    for (int g = 0; g < 8; ++g) {
      const int d = d0 + g + 8 * half;                 // C/D layout: row = g+8*half
      dst[((size_t)b * N_SP + n0 + col) * CQ + d] =
          static_cast<__bf16>(acc[g] + bias[d]);
    }
  } else {
    #pragma unroll
    for (int g = 0; g < 8; ++g) {
      const int d = d0 + g + 8 * half;
      vbuf[((size_t)b * C_CH + d) * N_SP + n0 + col] =
          static_cast<__bf16>(acc[g] + bias[d]);
    }
  }
}

// ---------------------------------------------------------------------------
// Kernel 2: fused flash-style attention + residual.
// One wave per (batch, 16-row Q block). Online softmax over 64-column KV
// steps (4 score WMMAs + 32 PV WMMAs per step); 16x256 f32 accumulator in
// VGPRs. Accumulator rescale / normalize done as v8f vector ops so the
// compiler can emit V_PK_MUL_F32 (halves VALU competing with the WMMA pipe).
// ---------------------------------------------------------------------------
__global__ __launch_bounds__(32)
void attn_kernel(const float* __restrict__ x,
                 const __bf16* __restrict__ qb,
                 const __bf16* __restrict__ kbuf,
                 const __bf16* __restrict__ vbuf,
                 const float* __restrict__ gamma,
                 float* __restrict__ out) {
  __shared__ __align__(16) __bf16 Pbuf[16 * 64];    // P tile bf16, 2KB
  __shared__ __align__(16) float  Obuf[C_CH * 16];  // [c][i] staging, 16KB

  const int lane = threadIdx.x & 31;
  const int half = lane >> 4;
  const int row  = lane & 15;
  const int col  = lane & 15;
  const int b    = blockIdx.x / (N_SP / 16);
  const int i0   = (blockIdx.x % (N_SP / 16)) * 16;

  // Q A-tile (persistent): qb is [B][N][32] bf16 -> two 16B loads per lane
  V16U qa;
  const __bf16* qr = qb + ((size_t)b * N_SP + i0 + row) * CQ;
  qa.u[0] = *reinterpret_cast<const uint4*>(qr + half * 16);
  qa.u[1] = *reinterpret_cast<const uint4*>(qr + half * 16 + 8);

  v8f acc[16];
  const v8f vzero = {0.f,0.f,0.f,0.f,0.f,0.f,0.f,0.f};
  #pragma unroll
  for (int t = 0; t < 16; ++t) acc[t] = vzero;
  float m[8], l[8];
  #pragma unroll
  for (int g = 0; g < 8; ++g) { m[g] = -3.0e38f; l[g] = 0.f; }

  const __bf16* kbase = kbuf + (size_t)b * N_SP * CQ;
  const __bf16* vbase = vbuf + (size_t)b * C_CH * N_SP;

  #pragma unroll 1
  for (int jb = 0; jb < N_SP / 64; ++jb) {
    const int j0 = jb * 64;

    // Prefetch next K block (64 rows x 64B = one line per row)
    if (jb + 1 < N_SP / 64) {
      __builtin_prefetch(kbase + (size_t)(j0 + 64 + lane) * CQ, 0, 1);
      __builtin_prefetch(kbase + (size_t)(j0 + 96 + lane) * CQ, 0, 1);
    }

    // 4 score WMMAs: S[i, j0+t*16+col]
    v8f s[4];
    #pragma unroll
    for (int t = 0; t < 4; ++t) {
      V16U kt;
      const __bf16* kp = kbase + (size_t)(j0 + t * 16 + col) * CQ + half * 16;
      kt.u[0] = *reinterpret_cast<const uint4*>(kp);
      kt.u[1] = *reinterpret_cast<const uint4*>(kp + 8);
      s[t] = __builtin_amdgcn_wmma_f32_16x16x32_bf16(false, qa.v, false, kt.v,
                                                     (short)0, vzero, false, false);
    }

    // Online softmax per row (row = g + 8*half; shfl masks <16 stay in-half)
    v8f alphav;
    #pragma unroll
    for (int g = 0; g < 8; ++g) {
      float mt = fmaxf(fmaxf(s[0][g], s[1][g]), fmaxf(s[2][g], s[3][g]));
      #pragma unroll
      for (int msk = 8; msk >= 1; msk >>= 1)
        mt = fmaxf(mt, __shfl_xor(mt, msk, 32));
      const float mn = fmaxf(m[g], mt);
      const float al = __expf(m[g] - mn);
      float p[4], rs = 0.f;
      #pragma unroll
      for (int t = 0; t < 4; ++t) { p[t] = __expf(s[t][g] - mn); rs += p[t]; }
      #pragma unroll
      for (int msk = 8; msk >= 1; msk >>= 1)
        rs += __shfl_xor(rs, msk, 32);
      l[g] = l[g] * al + rs;
      m[g] = mn;
      alphav[g] = al;
      const int r = g + 8 * half;
      #pragma unroll
      for (int t = 0; t < 4; ++t)
        Pbuf[r * 64 + t * 16 + col] = static_cast<__bf16>(p[t]);
    }
    // Rescale O accumulators: vector form -> V_PK_MUL_F32
    #pragma unroll
    for (int t = 0; t < 16; ++t) acc[t] *= alphav;
    __syncthreads();   // single wave: orders LDS P writes before A-layout reads

    // P A-tiles from LDS (C-layout -> A-layout transpose via LDS)
    V16U pa0, pa1;
    pa0.u[0] = *reinterpret_cast<const uint4*>(&Pbuf[row * 64 + half * 16]);
    pa0.u[1] = *reinterpret_cast<const uint4*>(&Pbuf[row * 64 + half * 16 + 8]);
    pa1.u[0] = *reinterpret_cast<const uint4*>(&Pbuf[row * 64 + 32 + half * 16]);
    pa1.u[1] = *reinterpret_cast<const uint4*>(&Pbuf[row * 64 + 40 + half * 16]);

    // O += P * V^T : 32 PV WMMAs (vbuf is [B][C][N] bf16, contiguous in j)
    #pragma unroll
    for (int t = 0; t < 16; ++t) {
      const __bf16* vp = vbase + (size_t)(t * 16 + col) * N_SP + j0 + half * 16;
      V16U vm0, vm1;
      vm0.u[0] = *reinterpret_cast<const uint4*>(vp);
      vm0.u[1] = *reinterpret_cast<const uint4*>(vp + 8);
      vm1.u[0] = *reinterpret_cast<const uint4*>(vp + 32);
      vm1.u[1] = *reinterpret_cast<const uint4*>(vp + 40);
      acc[t] = __builtin_amdgcn_wmma_f32_16x16x32_bf16(false, pa0.v, false, vm0.v,
                                                       (short)0, acc[t], false, false);
      acc[t] = __builtin_amdgcn_wmma_f32_16x16x32_bf16(false, pa1.v, false, vm1.v,
                                                       (short)0, acc[t], false, false);
    }
    __syncthreads();
  }

  // Finalize: normalize by l (vector mul), stage to LDS, fuse residual
  v8f invlv;
  #pragma unroll
  for (int g = 0; g < 8; ++g) invlv[g] = 1.0f / l[g];
  #pragma unroll
  for (int t = 0; t < 16; ++t) {
    const v8f o = acc[t] * invlv;
    #pragma unroll
    for (int g = 0; g < 8; ++g)
      Obuf[(t * 16 + col) * 16 + g + 8 * half] = o[g];
  }
  __syncthreads();
  const float gam = gamma[0];
  #pragma unroll
  for (int cc = 0; cc < 8; ++cc) {
    const int c = lane * 8 + cc;
    const float* xr   = x   + ((size_t)b * C_CH + c) * N_SP + i0;
    float*       orow = out + ((size_t)b * C_CH + c) * N_SP + i0;
    #pragma unroll
    for (int ii = 0; ii < 16; ++ii)
      orow[ii] = xr[ii] + gam * Obuf[c * 16 + ii];
  }
}

// ---------------------------------------------------------------------------
extern "C" void kernel_launch(void* const* d_in, const int* in_sizes, int n_in,
                              void* d_out, int out_size, void* d_ws, size_t ws_size,
                              hipStream_t stream) {
  const float* x     = (const float*)d_in[0];
  const float* wq    = (const float*)d_in[1];
  const float* bq    = (const float*)d_in[2];
  const float* wk    = (const float*)d_in[3];
  const float* bk    = (const float*)d_in[4];
  const float* wv    = (const float*)d_in[5];
  const float* bv    = (const float*)d_in[6];
  const float* gamma = (const float*)d_in[7];
  float*       out   = (float*)d_out;

  // Workspace: qb [B][N][32] bf16 (2MB), kb same (2MB), vb [B][C][N] bf16
  // (16MB), wbf [320x256] bf16 (160KB)
  __bf16* qb  = (__bf16*)d_ws;
  __bf16* kb  = qb + (size_t)BATCH * N_SP * CQ;
  __bf16* vb  = kb + (size_t)BATCH * N_SP * CQ;
  __bf16* wbf = vb + (size_t)BATCH * C_CH * N_SP;

  // Preconvert weights to bf16 (wq | wk | wv contiguous)
  cvt_kernel<<<(32 * 256 + 255) / 256, 256, 0, stream>>>(wq, wbf, 32 * 256);
  cvt_kernel<<<(32 * 256 + 255) / 256, 256, 0, stream>>>(wk, wbf + 32 * 256, 32 * 256);
  cvt_kernel<<<(256 * 256 + 255) / 256, 256, 0, stream>>>(wv, wbf + 64 * 256, 256 * 256);

  // One block per (batch, n-tile): 20 waves (2 q + 2 k + 16 v tiles)
  proj_kernel<<<BATCH * 256, 640, 0, stream>>>(
      x, wbf, bq, bk, bv, qb, kb, vb);

  // One wave per (batch, 16-row block): 8*256 = 2048 blocks of 32 threads
  attn_kernel<<<BATCH * (N_SP / 16), 32, 0, stream>>>(
      x, qb, kb, vb, gamma, out);
}
